// NeuroGauss4DPCI_72370198938042
// MI455X (gfx1250) — matrix operand order repaired and checked by reading, not compiled, not run on hardware
//
#include <hip/hip_runtime.h>
#include <hip/hip_bf16.h>

// ---------------------------------------------------------------------------
// Edge-gated GNN (3 layers) for MI455X (gfx1250, wave32, WMMA).
// - All GEMM-shaped work (node linear + per-edge MLP) on v_wmma_f32_16x16x32_bf16.
// - Edge gather uses CDNA5 async global->LDS loads (global_load_async_to_lds_b128,
//   ASYNCcnt) so the [x_i|x_j] tile lands in LDS without a VGPR round-trip.
// - Scores reduced with wave32 shuffles; aggregation via global_atomic_add_f32
//   (node tables are L2-resident: 30.7 MB << 192 MB L2).
// ---------------------------------------------------------------------------

typedef __attribute__((ext_vector_type(16))) __bf16 v16bf;
typedef __attribute__((ext_vector_type(8)))  __bf16 v8bf;
typedef __attribute__((ext_vector_type(8)))  float  v8f;

static __device__ __forceinline__ v16bf cat8(v8bf a, v8bf b) {
    return __builtin_shufflevector(a, b, 0,1,2,3,4,5,6,7,8,9,10,11,12,13,14,15);
}

// Load an A-fragment (16x32 bf16) for one lane from a row-major row pointer.
// Layout: lanes 0-15 hold K = {0..15}, lanes 16-31 K = {16..31} (matches B).
static __device__ __forceinline__ v16bf load_a_frag(const __bf16* rowp, int k0, int koff8) {
    v8bf lo = *(const v8bf*)(rowp + k0 + 2 * koff8);
    v8bf hi = *(const v8bf*)(rowp + k0 + 8 + 2 * koff8);
    return cat8(lo, hi);
}

// CDNA5 async copy: 16 bytes global -> LDS, tracked by ASYNCcnt.
static __device__ __forceinline__ void async_g2l_b128(uint32_t lds_byte_off,
                                                      const void* gaddr) {
    asm volatile("global_load_async_to_lds_b128 %0, %1, off"
                 :: "v"(lds_byte_off), "v"((uint64_t)(uintptr_t)gaddr)
                 : "memory");
}
static __device__ __forceinline__ void wait_asynccnt0() {
    asm volatile("s_wait_asynccnt 0x0" ::: "memory");
}

// ------------------------- small utility kernels ---------------------------

__global__ void pack_wT_bf16(const float* __restrict__ w, __bf16* __restrict__ wT,
                             int K, int Nn) {
    int i = blockIdx.x * 256 + threadIdx.x;
    if (i >= K * Nn) return;
    int n = i / K, k = i - n * K;
    wT[i] = (__bf16)w[(size_t)k * Nn + n];
}

__global__ void cvt_f32_bf16_k(const float* __restrict__ s, __bf16* __restrict__ d, int n) {
    int i = blockIdx.x * 256 + threadIdx.x;
    if (i < n) d[i] = (__bf16)s[i];
}

__global__ void zero_f32_k(float* __restrict__ p, int n) {
    int i = blockIdx.x * 256 + threadIdx.x;
    if (i < n) p[i] = 0.f;
}

__global__ void relu_cvt_k(const float* __restrict__ s, __bf16* __restrict__ d, int n) {
    int i = blockIdx.x * 256 + threadIdx.x;
    if (i < n) d[i] = (__bf16)fmaxf(s[i], 0.f);
}

// ----------------------- node linear: y = x @ w + b ------------------------
// x: [nrows, CI] bf16 row-major; wT: [CO][CI] bf16 (pre-transposed); y: bf16.
// One wave computes one 16x16 output tile; 8 waves per block.

template <int CI, int CO>
__global__ __launch_bounds__(256)
void linear_wmma(const __bf16* __restrict__ x, const __bf16* __restrict__ wT,
                 const float* __restrict__ bias, __bf16* __restrict__ y, int nrows) {
    const int lane = threadIdx.x & 31;
    const int wave = threadIdx.x >> 5;
    const int ntn  = CO / 16;
    const int tiles = (nrows / 16) * ntn;
    const int t = blockIdx.x * 8 + wave;
    if (t >= tiles) return;                       // uniform per wave
    const int mt = t / ntn, nt = t % ntn;
    const int lm    = lane & 15;
    const int half  = lane >> 4;
    const int koff8 = half * 8;                   // A-fragment K sub-offset
    const int bkoff = half * 16;                  // B-fragment K sub-offset
    const int arow = mt * 16 + lm;
    const int bcol = nt * 16 + lm;

    const float bv = bias[bcol];
    v8f acc = {bv, bv, bv, bv, bv, bv, bv, bv};
    const __bf16* ap = x  + (size_t)arow * CI;
    const __bf16* bp = wT + (size_t)bcol * CI + bkoff;
#pragma unroll
    for (int k0 = 0; k0 < CI; k0 += 32) {
        v16bf a = load_a_frag(ap, k0, koff8);
        v16bf b = *(const v16bf*)(bp + k0);
        acc = __builtin_amdgcn_wmma_f32_16x16x32_bf16(false, a, false, b,
                                                      (short)0, acc, false, false);
    }
    const int rbase = mt * 16 + 8 * half;
#pragma unroll
    for (int r = 0; r < 8; ++r)
        y[(size_t)(rbase + r) * CO + bcol] = (__bf16)acc[r];
}

// --------------------------- per-edge MLP + scatter ------------------------
// Block handles 32 edges. LDS-staged A = [x_i | x_j] (32 x 2C bf16, padded),
// filled by async global->LDS DMA. h = relu(A @ ws1 + b1) via WMMA;
// score = sigmoid(h . ws2 + b2); agg[dst] += score * x_j via f32 atomics.

template <int C>
__global__ __launch_bounds__(256)
void edge_wmma(const __bf16* __restrict__ xbf,   // [N, C]
               const int* __restrict__ ei,        // [2, E]
               const __bf16* __restrict__ ws1T,   // [C][2C]
               const float* __restrict__ bs1,     // [C]
               const float* __restrict__ ws2,     // [C]
               const float* __restrict__ bs2,     // [1]
               float* __restrict__ agg,           // [N, C]
               int nE, int nN) {
    constexpr int K2 = 2 * C;
    constexpr int AP = K2 + 16;   // padded LDS row (bf16 elems) -> no bank conflicts
    constexpr int HP = C + 8;
    __shared__ __align__(32) __bf16 At[32 * AP];
    __shared__ __align__(32) __bf16 Ht[32 * HP];
    __shared__ float ssc[32];
    __shared__ int   sdst[32], ssrc[32];

    const int tid = threadIdx.x;
    const int e0  = blockIdx.x * 32;

    if (tid < 32) {
        int e = e0 + tid, s = 0, d = -1;
        if (e < nE)            { s = ei[e]; d = ei[nE + e]; }
        else if (e < nE + nN)  { s = e - nE; d = s; }       // self loop
        ssrc[tid] = s; sdst[tid] = d;
    }
    __syncthreads();

    // gather rows: [x_i | x_j] straight into LDS via async DMA (16B chunks).
    // 32*segs is a multiple of 256 -> EXEC stays full through the issue loop.
    {
        constexpr int segs = K2 / 8;              // 8 bf16 per 16B chunk
        const uint32_t lbase = (uint32_t)(uintptr_t)(At);
#pragma unroll 1
        for (int t = tid; t < 32 * segs; t += 256) {
            const int row = t / segs, seg = t - row * segs;
            const int d = sdst[row];
            const int node = (seg < segs / 2) ? (d < 0 ? 0 : d) : ssrc[row];
            const int off  = (seg < segs / 2) ? seg * 8 : (seg - segs / 2) * 8;
            async_g2l_b128(lbase + (uint32_t)(row * AP + seg * 8) * 2,
                           xbf + (size_t)node * C + off);
        }
        wait_asynccnt0();                         // ASYNCcnt -> 0 before barrier
    }
    __syncthreads();

    // h = relu([x_i|x_j] @ ws1 + b1): 2 M-tiles x (C/16) N-tiles over 8 waves
    {
        const int lane = tid & 31, wave = tid >> 5;
        const int lm = lane & 15, half = lane >> 4;
        const int koff8 = half * 8, bkoff = half * 16;
        constexpr int NT = C / 16;
        for (int t = wave; t < 2 * NT; t += 8) {
            const int mt = t / NT, nt = t - mt * NT;
            const float bv = bs1[nt * 16 + lm];
            v8f acc = {bv, bv, bv, bv, bv, bv, bv, bv};
            const __bf16* ap = At + (mt * 16 + lm) * AP;
            const __bf16* bp = ws1T + (size_t)(nt * 16 + lm) * K2 + bkoff;
#pragma unroll
            for (int k0 = 0; k0 < K2; k0 += 32) {
                v16bf a = load_a_frag(ap, k0, koff8);
                v16bf b = *(const v16bf*)(bp + k0);
                acc = __builtin_amdgcn_wmma_f32_16x16x32_bf16(false, a, false, b,
                                                              (short)0, acc, false, false);
            }
            const int rbase = mt * 16 + 8 * half;
#pragma unroll
            for (int r = 0; r < 8; ++r)
                Ht[(rbase + r) * HP + nt * 16 + lm] = (__bf16)fmaxf(acc[r], 0.f);
        }
    }
    __syncthreads();

    // score = sigmoid(h . ws2 + b2): 8 threads per edge row + shuffle reduce
    {
        const int row = tid >> 3, part = tid & 7;
        float s = 0.f;
        for (int c = part; c < C; c += 8)
            s += (float)Ht[row * HP + c] * ws2[c];
        s += __shfl_down(s, 4, 8);
        s += __shfl_down(s, 2, 8);
        s += __shfl_down(s, 1, 8);
        if (part == 0) ssc[row] = 1.f / (1.f + __expf(-(s + bs2[0])));
    }
    __syncthreads();

    // scatter: agg[dst] += score * x_j
    for (int t = tid; t < 32 * C; t += 256) {
        const int row = t / C, c = t - row * C;
        const int d = sdst[row];
        if (d >= 0)
            atomicAdd(agg + (size_t)d * C + c, ssc[row] * (float)At[row * AP + C + c]);
    }
}

// ------------------------------- launcher ----------------------------------

static inline int cdiv(int a, int b) { return (a + b - 1) / b; }

extern "C" void kernel_launch(void* const* d_in, const int* in_sizes, int n_in,
                              void* d_out, int out_size, void* d_ws, size_t ws_size,
                              hipStream_t stream) {
    const int IN_C = 256, HID = 64, OUT_C = 256;
    const int N = in_sizes[0] / IN_C;       // 30000
    const int E = in_sizes[1] / 2;          // 480000
    const int EN = E + N;

    const float* x  = (const float*)d_in[0];
    const int*   ei = (const int*)d_in[1];
    const float *wl[3], *bl[3], *w1[3], *b1[3], *w2[3], *b2[3];
    for (int l = 0; l < 3; ++l) {
        wl[l] = (const float*)d_in[2 + 6 * l + 0];
        bl[l] = (const float*)d_in[2 + 6 * l + 1];
        w1[l] = (const float*)d_in[2 + 6 * l + 2];
        b1[l] = (const float*)d_in[2 + 6 * l + 3];
        w2[l] = (const float*)d_in[2 + 6 * l + 4];
        b2[l] = (const float*)d_in[2 + 6 * l + 5];
    }
    float* out = (float*)d_out;

    // workspace layout
    char* ws = (char*)d_ws;
    size_t off = 0;
    auto alloc = [&](size_t bytes) -> void* {
        off = (off + 255) & ~(size_t)255;
        void* p = ws + off; off += bytes; return p;
    };
    __bf16* bufA = (__bf16*)alloc((size_t)N * 256 * 2);   // layer input  (bf16)
    __bf16* bufB = (__bf16*)alloc((size_t)N * 256 * 2);   // linear out   (bf16)
    float*  aggb = (float*) alloc((size_t)N * HID * 4);   // agg for layers 1,2
    __bf16* wlT[3] = { (__bf16*)alloc(64 * 256 * 2),      // [CO][CI]
                       (__bf16*)alloc(64 * 64 * 2),
                       (__bf16*)alloc(256 * 64 * 2) };
    __bf16* w1T[3] = { (__bf16*)alloc(64 * 128 * 2),      // [C][2C]
                       (__bf16*)alloc(64 * 128 * 2),
                       (__bf16*)alloc(256 * 512 * 2) };
    (void)ws_size; (void)n_in; (void)out_size;

    // pack weights (fp32 -> bf16, transposed) and input x -> bf16
    const int ldims[3][2] = { {IN_C, HID}, {HID, HID}, {HID, OUT_C} };
    for (int l = 0; l < 3; ++l) {
        int K = ldims[l][0], Co = ldims[l][1];
        pack_wT_bf16<<<cdiv(K * Co, 256), 256, 0, stream>>>(wl[l], wlT[l], K, Co);
        pack_wT_bf16<<<cdiv(2 * Co * Co, 256), 256, 0, stream>>>(w1[l], w1T[l], 2 * Co, Co);
    }
    cvt_f32_bf16_k<<<cdiv(N * IN_C, 256), 256, 0, stream>>>(x, bufA, N * IN_C);

    const int eblocks = cdiv(EN, 32);

    // ---- layer 1: 256 -> 64 ----
    linear_wmma<256, 64><<<cdiv((N / 16) * (64 / 16), 8), 256, 0, stream>>>(
        bufA, wlT[0], bl[0], bufB, N);
    zero_f32_k<<<cdiv(N * 64, 256), 256, 0, stream>>>(aggb, N * 64);
    edge_wmma<64><<<eblocks, 256, 0, stream>>>(bufB, ei, w1T[0], b1[0], w2[0], b2[0],
                                               aggb, E, N);
    relu_cvt_k<<<cdiv(N * 64, 256), 256, 0, stream>>>(aggb, bufA, N * 64);

    // ---- layer 2: 64 -> 64 ----
    linear_wmma<64, 64><<<cdiv((N / 16) * (64 / 16), 8), 256, 0, stream>>>(
        bufA, wlT[1], bl[1], bufB, N);
    zero_f32_k<<<cdiv(N * 64, 256), 256, 0, stream>>>(aggb, N * 64);
    edge_wmma<64><<<eblocks, 256, 0, stream>>>(bufB, ei, w1T[1], b1[1], w2[1], b2[1],
                                               aggb, E, N);
    relu_cvt_k<<<cdiv(N * 64, 256), 256, 0, stream>>>(aggb, bufA, N * 64);

    // ---- layer 3: 64 -> 256, aggregate straight into d_out ----
    linear_wmma<64, 256><<<cdiv((N / 16) * (256 / 16), 8), 256, 0, stream>>>(
        bufA, wlT[2], bl[2], bufB, N);
    zero_f32_k<<<cdiv(N * 256, 256), 256, 0, stream>>>(out, N * 256);
    edge_wmma<256><<<eblocks, 256, 0, stream>>>(bufB, ei, w1T[2], b1[2], w2[2], b2[2],
                                                out, E, N);
}
